// NeighborAttention_16080357556864
// MI455X (gfx1250) — matrix operand order repaired and checked by reading, not compiled, not run on hardware
//
#include <hip/hip_runtime.h>
#include <hip/hip_fp16.h>

typedef __attribute__((ext_vector_type(16))) _Float16 v16h;
typedef __attribute__((ext_vector_type(8)))  float    v8f;

// ---------------------------------------------------------------------------
// Fragment address helpers (CDNA5 WMMA 16x16x32 f16 layouts, ISA 7.12.2)
// ---------------------------------------------------------------------------
// A-matrix 16x32 (MxK), element (m, kl) with kl in [0,32):
//   lanes 0-15 hold kk(=kl%16) in 0..7, lanes 16-31 hold kk 8..15,
//   VGPRs 0-3 (halves 0..7) hold K-group 0, VGPRs 4-7 (halves 8..15) group 1.
__device__ __forceinline__ int a_frag_addr(int m, int kl) {
  int g  = kl >> 4;           // K group (0,1)
  int kk = kl & 15;
  int lane = m + ((kk >> 3) << 4);
  int h    = (g << 3) + (kk & 7);
  return lane * 16 + h;       // index into 512-halfword fragment block
}
// B-matrix 32x16 (KxN): lane = n + 16*((kl/8)%2), half = 8*(kl/16) + kl%8 (mirrors the
// 64x16 sparse-B layout in the ISA, scaled to K=32).
// C/D f32 16x16: lane = n + 16*(m/8), vgpr r = m%8.

// ---------------------------------------------------------------------------
// Atomics
// ---------------------------------------------------------------------------
__device__ __forceinline__ void atomicMaxFloat(float* addr, float val) {
  unsigned int* ua = (unsigned int*)addr;
  unsigned int old = *ua;
  while (__uint_as_float(old) < val) {
    unsigned int assumed = old;
    old = atomicCAS(ua, assumed, __float_as_uint(val));
    if (old == assumed) break;
  }
}
__device__ __forceinline__ void atomicAddF(float* addr, float val) {
  __hip_atomic_fetch_add(addr, val, __ATOMIC_RELAXED, __HIP_MEMORY_SCOPE_AGENT);
}

// ---------------------------------------------------------------------------
// WMMA tile helpers
// ---------------------------------------------------------------------------
// Two M-tiles share each B fragment (weight traffic halved). Fully unrolled.
template<int NKB>
__device__ __forceinline__ void mm_tile2(const _Float16* __restrict__ in0,
                                         const _Float16* __restrict__ in1,
                                         const _Float16* __restrict__ wp,
                                         int lane, v8f& acc0, v8f& acc1) {
#pragma unroll
  for (int kb = 0; kb < NKB; ++kb) {
    const int o = kb * 512 + lane * 16;
    if (kb + 1 < NKB) __builtin_prefetch(wp + (kb + 1) * 512 + lane * 16, 0, 3);
    v16h b  = *(const v16h*)(wp  + o);
    v16h a0 = *(const v16h*)(in0 + o);
    v16h a1 = *(const v16h*)(in1 + o);
    acc0 = __builtin_amdgcn_wmma_f32_16x16x32_f16(false, a0, false, b, (short)0, acc0, false, false);
    acc1 = __builtin_amdgcn_wmma_f32_16x16x32_f16(false, a1, false, b, (short)0, acc1, false, false);
  }
}

// Epilogue: bias + activation, scatter C-layout back into A-fragment layout
// (feeds the next layer). ACT: 0=none, 1=relu, 2=exact gelu.
template<int ACT>
__device__ __forceinline__ void store_act_frag(_Float16* out, int nkbOut, int mt, int nt,
                                               int lane, const v8f& acc, const float* bias) {
  float bv  = bias[nt * 16 + (lane & 15)];
  int mbase = (lane >> 4) << 3;
  int kg    = nt * 16 + (lane & 15);   // this column becomes a K index next layer
  int kb2   = kg >> 5, kl = kg & 31;
  _Float16* base = out + (mt * nkbOut + kb2) * 512;
#pragma unroll
  for (int r = 0; r < 8; ++r) {
    float x = acc[r] + bv;
    if (ACT == 1) x = fmaxf(x, 0.f);
    if (ACT == 2) x = 0.5f * x * (1.f + erff(x * 0.70710678118654752f));
    base[a_frag_addr(mbase + r, kl)] = (_Float16)x;
  }
}

template<int ACT, int NKBIN>
__device__ __forceinline__ void layer_lds(const _Float16* in, int mtStrideIn,
                                          _Float16* out, int nkbOut,
                                          const _Float16* __restrict__ Wp, const float* bias,
                                          int wave, int lane) {
#pragma unroll
  for (int t = 0; t < 2; ++t) {
    int nt = wave * 2 + t;
    v8f a0, a1;
#pragma unroll
    for (int r = 0; r < 8; ++r) { a0[r] = 0.f; a1[r] = 0.f; }
    mm_tile2<NKBIN>(in, in + mtStrideIn, Wp + nt * NKBIN * 512, lane, a0, a1);
    store_act_frag<ACT>(out, nkbOut, 0, nt, lane, a0, bias);
    store_act_frag<ACT>(out, nkbOut, 1, nt, lane, a1, bias);
  }
}

// ---------------------------------------------------------------------------
// Weight packing: W[outF][inF] fp32 -> B-fragment-layout fp16
// ---------------------------------------------------------------------------
__global__ void pack_w_kernel(const float* __restrict__ W, _Float16* __restrict__ Wp,
                              int inF, int outReal, int total) {
  int i = blockIdx.x * 256 + threadIdx.x;
  if (i >= total) return;
  int nkb  = inF >> 5;
  int nt   = i / (nkb * 512);
  int rem  = i - nt * (nkb * 512);
  int kb   = rem >> 9;
  int q    = rem & 511;
  int lane = q >> 4, h = q & 15;
  int ncol = lane & 15;
  int kl   = ((h >> 3) << 4) + ((lane >> 4) << 3) + (h & 7);
  int orow = nt * 16 + ncol;
  int k    = (kb << 5) + kl;
  float v  = (orow < outReal) ? W[(long long)orow * inF + k] : 0.f;
  Wp[i] = (_Float16)v;
}

__global__ void init_nodes_kernel(float* node_max, float* node_sum, float* node_agg, int Nn) {
  int i = blockIdx.x * 256 + threadIdx.x;
  if (i < Nn * 8) { node_max[i] = -3.0e38f; node_sum[i] = 0.f; }
  if (i < Nn * 256) node_agg[i] = 0.f;
}

// ---------------------------------------------------------------------------
// Edge MLP kernel: 32 edges / block, everything through WMMA
// ---------------------------------------------------------------------------
__global__ __launch_bounds__(256)
void edge_mlp_kernel(const float* __restrict__ src_na, const float* __restrict__ dst_na,
                     const float* __restrict__ ea, const long long* __restrict__ edge_idx,
                     const _Float16* __restrict__ Wb1p, const float* __restrict__ bb1,
                     const _Float16* __restrict__ Wb2p, const float* __restrict__ bb2,
                     const _Float16* __restrict__ Wb3p, const float* __restrict__ bb3,
                     const _Float16* __restrict__ Wv1p, const float* __restrict__ bv1,
                     const _Float16* __restrict__ Wv2p, const float* __restrict__ bv2,
                     const _Float16* __restrict__ Wv3p, const float* __restrict__ bv3,
                     float* __restrict__ edge_logits, _Float16* __restrict__ edge_V,
                     float* __restrict__ node_max, int E) {
  __shared__ _Float16 ldsX[32 * 768];   // [mt][24 kb][512] : concat(s, ea, t) frags
  __shared__ _Float16 ldsA[32 * 256];   // [mt][8 kb][512]  : activation ping
  __shared__ _Float16 ldsB[32 * 256];   //                  : activation pong
  __shared__ int ldsDst[32];
  __shared__ int ldsSrc[32];

  const int e0 = blockIdx.x * 32;
  const int tid = threadIdx.x;
  const int wave = tid >> 5, lane = tid & 31;

  // ---- indices ----
  if (tid < 32) {
    int e = e0 + tid;
    ldsDst[tid] = (e < E) ? (int)edge_idx[e] : 0;
  } else if (tid < 64) {
    int t = tid - 32, e = e0 + t;
    ldsSrc[t] = (e < E) ? (int)edge_idx[(long long)E + e] : 0;
  }
  __syncthreads();

  // ---- gather + fp32->fp16 directly into A-fragment layout ----
  for (int idx = tid; idx < 32 * 768; idx += 256) {
    int r = idx / 768, c = idx - r * 768;
    int e = e0 + r;
    float v = 0.f;
    if (e < E) {
      if (c < 256)       v = src_na[(long long)ldsSrc[r] * 256 + c];
      else if (c < 512)  v = ea[(long long)e * 256 + (c - 256)];
      else               v = dst_na[(long long)ldsDst[r] * 256 + (c - 512)];
    }
    int mt = r >> 4, m = r & 15, kb = c >> 5, kl = c & 31;
    ldsX[(mt * 24 + kb) * 512 + a_frag_addr(m, kl)] = (_Float16)v;
  }
  __syncthreads();

  // ---- bias MLP: 768 -> 256 (relu) -> 256 (relu) -> 8 logits ----
  layer_lds<1, 24>(ldsX, 24 * 512, ldsA, 8, Wb1p, bb1, wave, lane);
  __syncthreads();
  layer_lds<1, 8>(ldsA, 8 * 512, ldsB, 8, Wb2p, bb2, wave, lane);
  __syncthreads();

  if (wave == 0) {   // final 256 -> 8 (padded to 16 cols), fused scale + segment-max
    v8f a0, a1;
#pragma unroll
    for (int r = 0; r < 8; ++r) { a0[r] = 0.f; a1[r] = 0.f; }
    mm_tile2<8>(ldsB, ldsB + 8 * 512, Wb3p, lane, a0, a1);
    int col = lane & 15, mbase = (lane >> 4) << 3;
    if (col < 8) {
      float bv = bb3[col];
#pragma unroll
      for (int mt = 0; mt < 2; ++mt) {
        v8f acc = mt ? a1 : a0;
#pragma unroll
        for (int r = 0; r < 8; ++r) {
          int row = e0 + mt * 16 + mbase + r;
          if (row < E) {
            float lg = (acc[r] + bv) * 0.17677669529663687f;  // 1/sqrt(d), d=32
            edge_logits[(long long)row * 8 + col] = lg;
            atomicMaxFloat(node_max + (long long)ldsDst[mt * 16 + mbase + r] * 8 + col, lg);
          }
        }
      }
    }
  }

  // ---- value MLP: 512 -> 256 (gelu) -> 256 (gelu) -> 256 ----
  layer_lds<2, 16>(ldsX, 24 * 512, ldsA, 8, Wv1p, bv1, wave, lane);
  __syncthreads();                       // also fences wave0's read of ldsB above
  layer_lds<2, 8>(ldsA, 8 * 512, ldsB, 8, Wv2p, bv2, wave, lane);
  __syncthreads();

#pragma unroll
  for (int t = 0; t < 2; ++t) {          // final linear, store V as fp16
    int nt = wave * 2 + t;
    v8f a0, a1;
#pragma unroll
    for (int r = 0; r < 8; ++r) { a0[r] = 0.f; a1[r] = 0.f; }
    mm_tile2<8>(ldsB, ldsB + 8 * 512, Wv3p + nt * 8 * 512, lane, a0, a1);
    int col = lane & 15, mbase = (lane >> 4) << 3;
    float bv = bv3[nt * 16 + col];
#pragma unroll
    for (int mt = 0; mt < 2; ++mt) {
      v8f acc = mt ? a1 : a0;
#pragma unroll
      for (int r = 0; r < 8; ++r) {
        int row = e0 + mt * 16 + mbase + r;
        if (row < E)
          edge_V[(long long)row * 256 + nt * 16 + col] = (_Float16)(acc[r] + bv);
      }
    }
  }
}

// ---------------------------------------------------------------------------
// Softmax numerator + denominator
// ---------------------------------------------------------------------------
__global__ void ex_sum_kernel(const float* __restrict__ edge_logits,
                              const long long* __restrict__ edge_idx,
                              const float* __restrict__ node_max,
                              float* __restrict__ edge_ex, float* __restrict__ node_sum, int E) {
  int i = blockIdx.x * 256 + threadIdx.x;
  if (i >= E * 8) return;
  int e = i >> 3, hh = i & 7;
  int dst = (int)edge_idx[e];
  float ex = __expf(edge_logits[i] - node_max[dst * 8 + hh]);
  edge_ex[i] = ex;
  atomicAddF(node_sum + dst * 8 + hh, ex);
}

// ---------------------------------------------------------------------------
// attend * V scatter-add (8 channels per thread)
// ---------------------------------------------------------------------------
__global__ void aggregate_kernel(const float* __restrict__ edge_ex,
                                 const _Float16* __restrict__ edge_V,
                                 const long long* __restrict__ edge_idx,
                                 const float* __restrict__ node_sum,
                                 float* __restrict__ node_agg, int E) {
  int i = blockIdx.x * 256 + threadIdx.x;
  if (i >= E * 32) return;
  int e = i >> 5, c0 = (i & 31) * 8;
  int dst = (int)edge_idx[e];
  int hh = c0 >> 5;                                     // head = channel/32
  float at = edge_ex[e * 8 + hh] / node_sum[dst * 8 + hh];
  const _Float16* vp = edge_V + (long long)e * 256 + c0;
  float* op = node_agg + (long long)dst * 256 + c0;
#pragma unroll
  for (int j = 0; j < 8; ++j) atomicAddF(op + j, at * (float)vp[j]);
}

// ---------------------------------------------------------------------------
// out = node_agg @ WO.T via WMMA, 16 nodes / block
// ---------------------------------------------------------------------------
__global__ __launch_bounds__(256)
void out_gemm_kernel(const float* __restrict__ node_agg, const _Float16* __restrict__ WOp,
                     float* __restrict__ out, int Nn) {
  __shared__ _Float16 ldsA[16 * 256];
  const int n0 = blockIdx.x * 16;
  const int tid = threadIdx.x, wave = tid >> 5, lane = tid & 31;
  for (int idx = tid; idx < 16 * 256; idx += 256) {
    int m = idx >> 8, c = idx & 255;
    float v = (n0 + m < Nn) ? node_agg[(long long)(n0 + m) * 256 + c] : 0.f;
    ldsA[(c >> 5) * 512 + a_frag_addr(m, c & 31)] = (_Float16)v;
  }
  __syncthreads();
#pragma unroll
  for (int t = 0; t < 2; ++t) {
    int nt = wave * 2 + t;
    v8f acc;
#pragma unroll
    for (int r = 0; r < 8; ++r) acc[r] = 0.f;
#pragma unroll
    for (int kb = 0; kb < 8; ++kb) {
      v16h a = *(const v16h*)(ldsA + kb * 512 + lane * 16);
      v16h b = *(const v16h*)(WOp + (nt * 8 + kb) * 512 + lane * 16);
      acc = __builtin_amdgcn_wmma_f32_16x16x32_f16(false, a, false, b, (short)0, acc, false, false);
    }
    int col = lane & 15, mbase = (lane >> 4) << 3;
#pragma unroll
    for (int r = 0; r < 8; ++r) {
      int row = n0 + mbase + r;
      if (row < Nn) out[(long long)row * 256 + nt * 16 + col] = acc[r];
    }
  }
}

// ---------------------------------------------------------------------------
// Host launch
// ---------------------------------------------------------------------------
extern "C" void kernel_launch(void* const* d_in, const int* in_sizes, int n_in,
                              void* d_out, int out_size, void* d_ws, size_t ws_size,
                              hipStream_t stream) {
  (void)n_in; (void)out_size; (void)ws_size;
  const float* src_na = (const float*)d_in[0];
  const float* dst_na = (const float*)d_in[1];
  const float* ea     = (const float*)d_in[2];
  const long long* edge_idx = (const long long*)d_in[3];
  const float* Wv1 = (const float*)d_in[4];  const float* bv1 = (const float*)d_in[5];
  const float* Wv2 = (const float*)d_in[6];  const float* bv2 = (const float*)d_in[7];
  const float* Wv3 = (const float*)d_in[8];  const float* bv3 = (const float*)d_in[9];
  const float* Wb1 = (const float*)d_in[10]; const float* bb1 = (const float*)d_in[11];
  const float* Wb2 = (const float*)d_in[12]; const float* bb2 = (const float*)d_in[13];
  const float* Wb3 = (const float*)d_in[14]; const float* bb3 = (const float*)d_in[15];
  const float* WO  = (const float*)d_in[16];

  const int Nn = in_sizes[0] / 256;
  const int E  = in_sizes[2] / 256;

  char* ws = (char*)d_ws;
  size_t off = 0;
  auto alloc = [&](size_t bytes) -> void* {
    void* p = ws + off;
    off = (off + bytes + 255) & ~(size_t)255;
    return p;
  };
  _Float16* Wv1p = (_Float16*)alloc((size_t)256 * 512 * 2);
  _Float16* Wv2p = (_Float16*)alloc((size_t)256 * 256 * 2);
  _Float16* Wv3p = (_Float16*)alloc((size_t)256 * 256 * 2);
  _Float16* Wb1p = (_Float16*)alloc((size_t)256 * 768 * 2);
  _Float16* Wb2p = (_Float16*)alloc((size_t)256 * 256 * 2);
  _Float16* Wb3p = (_Float16*)alloc((size_t)16  * 256 * 2);   // 8 real + 8 zero cols
  _Float16* WOp  = (_Float16*)alloc((size_t)256 * 256 * 2);
  float* node_max = (float*)alloc((size_t)Nn * 8 * 4);
  float* node_sum = (float*)alloc((size_t)Nn * 8 * 4);
  float* node_agg = (float*)alloc((size_t)Nn * 256 * 4);
  float* edge_logits = (float*)alloc((size_t)E * 8 * 4);
  float* edge_ex     = (float*)alloc((size_t)E * 8 * 4);
  _Float16* edge_V   = (_Float16*)alloc((size_t)E * 256 * 2);

  auto packW = [&](const float* W, _Float16* Wp, int outF, int inF, int outReal) {
    int total = (outF / 16) * (inF / 32) * 512;
    pack_w_kernel<<<(total + 255) / 256, 256, 0, stream>>>(W, Wp, inF, outReal, total);
  };
  packW(Wv1, Wv1p, 256, 512, 256);
  packW(Wv2, Wv2p, 256, 256, 256);
  packW(Wv3, Wv3p, 256, 256, 256);
  packW(Wb1, Wb1p, 256, 768, 256);
  packW(Wb2, Wb2p, 256, 256, 256);
  packW(Wb3, Wb3p, 16,  256, 8);
  packW(WO,  WOp,  256, 256, 256);

  init_nodes_kernel<<<(Nn * 256 + 255) / 256, 256, 0, stream>>>(node_max, node_sum, node_agg, Nn);

  edge_mlp_kernel<<<(E + 31) / 32, 256, 0, stream>>>(
      src_na, dst_na, ea, edge_idx,
      Wb1p, bb1, Wb2p, bb2, Wb3p, bb3,
      Wv1p, bv1, Wv2p, bv2, Wv3p, bv3,
      edge_logits, edge_V, node_max, E);

  ex_sum_kernel<<<(E * 8 + 255) / 256, 256, 0, stream>>>(
      edge_logits, edge_idx, node_max, edge_ex, node_sum, E);

  aggregate_kernel<<<(E * 32 + 255) / 256, 256, 0, stream>>>(
      edge_ex, edge_V, edge_idx, node_sum, node_agg, E);

  out_gemm_kernel<<<(Nn + 15) / 16, 256, 0, stream>>>(node_agg, WOp, (float*)d_out, Nn);
}
